// KNN_3341484556526
// MI455X (gfx1250) — matrix operand order repaired and checked by reading, not compiled, not run on hardware
//
#include <hip/hip_runtime.h>
#include <hip/hip_bf16.h>

// ---------------------------------------------------------------------------
// Fused KNN for MI455X (gfx1250, wave32, WMMA).
//   d2[i,j] = ||test_i||^2 + ||train_j||^2 - 2 * <test_i, train_j>
// GEMM in f16 via v_wmma_f32_16x16x32_f16 (f32 accumulate), norms in f32.
// Each wave owns 16 test rows (A tile resident in 128 VGPRs) and streams all
// 40000 train rows from L2 (train_f16 = 41MB fits the 192MB L2). Two train
// column tiles per iteration -> two independent WMMA accumulation chains for
// better matrix-pipe and load overlap. Top-32 selection fused per wave.
// ---------------------------------------------------------------------------

#define N_TRAIN   40000
#define N_TEST    10000
#define DIM       512
#define N_CH      64
#define K_NN      32
#define WAVES_PB  8
#define ROWS_PW   16
#define TEST_PAD  10112            // 79 blocks * 8 waves * 16 rows
#define N_ITERS   (N_TRAIN / 32)   // 1250 iterations of 2 column tiles
#define K_STEPS   (DIM / 32)       // 16 wmma k-steps

typedef __attribute__((ext_vector_type(16))) _Float16 v16h;
typedef __attribute__((ext_vector_type(8)))  float    v8f;

// ---------- prep: f32 -> f16 + squared row norms (one wave per row) --------
__global__ __launch_bounds__(256)
void knn_prep(const float* __restrict__ src, _Float16* __restrict__ dst,
              float* __restrict__ sq, int rows, int validRows) {
    const int w    = threadIdx.x >> 5;
    const int lane = threadIdx.x & 31;
    const int row  = blockIdx.x * WAVES_PB + w;
    if (row >= rows) return;

    const float* s = src + (size_t)row * DIM;
    _Float16*    d = dst + (size_t)row * DIM;
    float sum = 0.0f;
    if (row < validRows) {
        #pragma unroll
        for (int t = 0; t < DIM / 32; ++t) {
            float v = s[t * 32 + lane];
            sum += v * v;
            d[t * 32 + lane] = (_Float16)v;
        }
    } else { // zero pad rows so WMMA tiles read valid data
        #pragma unroll
        for (int t = 0; t < DIM / 32; ++t) d[t * 32 + lane] = (_Float16)0.0f;
    }
    #pragma unroll
    for (int o = 16; o >= 1; o >>= 1) sum += __shfl_xor(sum, o, 32);
    if (lane == 0) sq[row] = sum;
}

// ---------- main: fused GEMM + streaming top-32 + label gather -------------
__global__ __launch_bounds__(256)
void knn_main(const _Float16* __restrict__ testH,
              const _Float16* __restrict__ trainH,
              const float* __restrict__ testSq,
              const float* __restrict__ trainSq,
              const float* __restrict__ trainY,
              float* __restrict__ out) {
    __shared__ float s_dl[WAVES_PB][ROWS_PW * K_NN];
    __shared__ int   s_il[WAVES_PB][ROWS_PW * K_NN];
    __shared__ float s_thr[WAVES_PB][ROWS_PW];
    __shared__ int   s_mp[WAVES_PB][ROWS_PW];

    const int w     = threadIdx.x >> 5;
    const int lane  = threadIdx.x & 31;
    const int half  = lane >> 4;   // 0: lanes 0-15, 1: lanes 16-31
    const int l16   = lane & 15;
    const int wRow0 = (blockIdx.x * WAVES_PB + w) * ROWS_PW; // wave's test rows

    volatile float* dlV  = s_dl[w];
    volatile int*   ilV  = s_il[w];
    volatile float* thrV = s_thr[w];
    volatile int*   mpV  = s_mp[w];

    // init wave-private top-k state
    #pragma unroll
    for (int r = 0; r < ROWS_PW; ++r) dlV[r * K_NN + lane] = 3.0e38f;
    #pragma unroll
    for (int r = 0; r < ROWS_PW; ++r) ilV[r * K_NN + lane] = 0;
    if (lane < ROWS_PW) { thrV[lane] = 3.0e38f; mpV[lane] = 0; }

    // Preload A tile (16 test rows x 512 f16) into 16 v16h fragments.
    // ISA 16-bit A 16x32 layout: lanes 0-15 -> M=0..15 with K {0..7,16..23},
    // lanes 16-31 -> same M with K {8..15,24..31}.
    v16h aFrag[K_STEPS];
    {
        const _Float16* tb = testH + (size_t)(wRow0 + l16) * DIM;
        #pragma unroll
        for (int s = 0; s < K_STEPS; ++s) {
            const _Float16* p = tb + s * 32 + half * 8;
            #pragma unroll
            for (int j = 0; j < 8; ++j) {
                aFrag[s][j]     = p[j];
                aFrag[s][j + 8] = p[j + 16];
            }
        }
    }

    // Per-lane test norms for the 8 C rows this lane holds (M = i + half*8).
    float tSq[8];
    #pragma unroll
    for (int i = 0; i < 8; ++i) tSq[i] = testSq[wRow0 + i + half * 8];

    for (int it = 0; it < N_ITERS; ++it) {
        const int colBase = it * 32;
        const int col0    = colBase + l16;
        const int col1    = col0 + 16;

        // B 32x16 layout: lane = train row (column of B), same K order as A:
        // lanes 0-15 K {0..7,16..23}, lanes 16-31 K {8..15,24..31}.
        const _Float16* bb0 = trainH + (size_t)col0 * DIM + half * 8;
        const _Float16* bb1 = trainH + (size_t)col1 * DIM + half * 8;

        v8f acc0 = {0.f, 0.f, 0.f, 0.f, 0.f, 0.f, 0.f, 0.f};
        v8f acc1 = {0.f, 0.f, 0.f, 0.f, 0.f, 0.f, 0.f, 0.f};
        #pragma unroll
        for (int s = 0; s < K_STEPS; ++s) {
            v16h b0, b1;
            const _Float16* p0 = bb0 + s * 32;
            const _Float16* p1 = bb1 + s * 32;
            #pragma unroll
            for (int j = 0; j < 8; ++j) { b0[j] = p0[j]; b0[j + 8] = p0[j + 16]; }
            #pragma unroll
            for (int j = 0; j < 8; ++j) { b1[j] = p1[j]; b1[j + 8] = p1[j + 16]; }
            acc0 = __builtin_amdgcn_wmma_f32_16x16x32_f16(
                false, aFrag[s], false, b0, (short)0, acc0, false, false);
            acc1 = __builtin_amdgcn_wmma_f32_16x16x32_f16(
                false, aFrag[s], false, b1, (short)0, acc1, false, false);
        }

        // ---- streaming top-32 over the 2 x (16x16) distance tiles ----
        #pragma unroll
        for (int t = 0; t < 2; ++t) {
            const int   col = (t == 0) ? col0 : col1;
            const float cSq = trainSq[col];
            #pragma unroll
            for (int i = 0; i < 8; ++i) {
                const int   row = i + half * 8;         // local test row
                const float dot = (t == 0) ? acc0[i] : acc1[i];
                const float d   = tSq[i] + cSq - 2.0f * dot;
                bool pend = d < thrV[row];
                while (true) {
                    unsigned m = __builtin_amdgcn_ballot_w32(pend);
                    if (m == 0u) break;
                    unsigned leader = (unsigned)__builtin_ctz(m);
                    if ((unsigned)lane == leader) {
                        if (d < thrV[row]) { // re-check: threshold may have moved
                            int p = mpV[row];
                            dlV[row * K_NN + p] = d;
                            ilV[row * K_NN + p] = col;
                            float mx = dlV[row * K_NN + 0]; int mi = 0;
                            #pragma unroll 4
                            for (int j = 1; j < K_NN; ++j) {
                                float v = dlV[row * K_NN + j];
                                if (v > mx) { mx = v; mi = j; }
                            }
                            thrV[row] = mx; mpV[row] = mi;
                        }
                        pend = false;
                    }
                }
            }
        }
    }

    // Gather labels: mean over 32 neighbors, 64 channels (2 per lane).
    for (int r = 0; r < ROWS_PW; ++r) {
        const int gRow = wRow0 + r;
        if (gRow >= N_TEST) continue;   // uniform per wave
        float a0 = 0.0f, a1 = 0.0f;
        for (int j = 0; j < K_NN; ++j) {
            const int idx = ilV[r * K_NN + j];          // scalar broadcast
            const float* yr = trainY + (size_t)idx * N_CH;
            a0 += yr[lane];
            a1 += yr[lane + 32];
        }
        out[(size_t)gRow * N_CH + lane]      = a0 * (1.0f / K_NN);
        out[(size_t)gRow * N_CH + lane + 32] = a1 * (1.0f / K_NN);
    }
}

// ---------------------------------------------------------------------------
extern "C" void kernel_launch(void* const* d_in, const int* in_sizes, int n_in,
                              void* d_out, int out_size, void* d_ws, size_t ws_size,
                              hipStream_t stream) {
    const float* train_x = (const float*)d_in[0];  // [40000,512]
    const float* train_y = (const float*)d_in[1];  // [40000,64]
    const float* test_x  = (const float*)d_in[2];  // [10000,512]
    float*       out     = (float*)d_out;          // [10000,64]

    char* ws = (char*)d_ws;
    const size_t OFF_TRAINH = 0;                                    // 40,960,000 B
    const size_t OFF_TESTH  = OFF_TRAINH + (size_t)N_TRAIN * DIM * 2;
    const size_t OFF_TRSQ   = OFF_TESTH  + (size_t)TEST_PAD * DIM * 2;
    const size_t OFF_TESQ   = OFF_TRSQ   + (size_t)N_TRAIN * 4;

    _Float16* trainH = (_Float16*)(ws + OFF_TRAINH);
    _Float16* testH  = (_Float16*)(ws + OFF_TESTH);
    float*    trainSq = (float*)(ws + OFF_TRSQ);
    float*    testSq  = (float*)(ws + OFF_TESQ);

    knn_prep<<<N_TRAIN / WAVES_PB, 256, 0, stream>>>(train_x, trainH, trainSq,
                                                     N_TRAIN, N_TRAIN);
    knn_prep<<<TEST_PAD / WAVES_PB, 256, 0, stream>>>(test_x, testH, testSq,
                                                      TEST_PAD, N_TEST);
    knn_main<<<TEST_PAD / (WAVES_PB * ROWS_PW), 256, 0, stream>>>(
        testH, trainH, testSq, trainSq, train_y, out);
}
